// MyGCNModel_62199716381134
// MI455X (gfx1250) — matrix operand order repaired and checked by reading, not compiled, not run on hardware
//
#include <hip/hip_runtime.h>
#include <hip/hip_bf16.h>

#define DD 2048   // feature dim, fixed by the module

typedef __attribute__((ext_vector_type(16))) __bf16 v16bf;
typedef __attribute__((ext_vector_type(8)))  float  v8f;

// ---------- helpers ----------
__device__ __forceinline__ unsigned short f32_to_bf16_rne(float f) {
    unsigned int u = __float_as_uint(f);
    unsigned int r = u + 0x7FFFu + ((u >> 16) & 1u);   // round-to-nearest-even
    return (unsigned short)(r >> 16);
}

// ---------- degree: deg[dst[e]] += 1 ----------
__global__ void deg_kernel(const int* __restrict__ dst, float* __restrict__ deg, int E) {
    int t = blockIdx.x * blockDim.x + threadIdx.x;
    if (t < E) unsafeAtomicAdd(deg + dst[t], 1.0f);
}

// ---------- weight convert: wt[s][n][k] = bf16( (s&1 ? Wn : Wr)[s>>1][k][n] ) ----------
__global__ void wconv_kernel(const float* __restrict__ Wr, const float* __restrict__ Wn,
                             unsigned short* __restrict__ wt) {
    int t = blockIdx.x * blockDim.x + threadIdx.x;
    if (t >= 8 * DD * DD) return;
    int s   = t / (DD * DD);
    int idx = t - s * (DD * DD);
    int n   = idx / DD;
    int k   = idx - n * DD;
    int c   = s >> 1;
    const float* W = (s & 1) ? Wn : Wr;
    float v = W[(size_t)c * DD * DD + (size_t)k * DD + n];
    wt[t] = f32_to_bf16_rne(v);
}

// ---------- edge scatter: agg[dst] += in[src]  (4 floats / thread) ----------
__global__ void scatter_kernel(const float* __restrict__ in, const int* __restrict__ src,
                               const int* __restrict__ dst, float* __restrict__ agg, int E) {
    const int per = DD / 4;
    int t = blockIdx.x * blockDim.x + threadIdx.x;
    int e = t / per;
    if (e >= E) return;
    int c = (t - e * per) * 4;
    int s = src[e], d = dst[e];
    float4 v = *(const float4*)(in + (size_t)s * DD + c);
    float* p = agg + (size_t)d * DD + c;
    unsafeAtomicAdd(p + 0, v.x);
    unsafeAtomicAdd(p + 1, v.y);
    unsafeAtomicAdd(p + 2, v.z);
    unsafeAtomicAdd(p + 3, v.w);
}

// ---------- normalize agg by deg, convert in & agg to bf16 (4 elems / thread) ----------
__global__ void normconv_kernel(const float* __restrict__ in, const float* __restrict__ agg,
                                const float* __restrict__ deg,
                                unsigned short* __restrict__ xb, unsigned short* __restrict__ aggb,
                                int n) {
    int t = blockIdx.x * blockDim.x + threadIdx.x;
    size_t i4 = (size_t)t * 4;
    if (i4 >= (size_t)n * DD) return;
    int row = (int)(i4 / DD);
    float inv = 1.0f / fmaxf(deg[row], 1.0f);
    float4 xv = *(const float4*)(in + i4);
    float4 av = *(const float4*)(agg + i4);
    ushort4 xo, ao;
    xo.x = f32_to_bf16_rne(xv.x); xo.y = f32_to_bf16_rne(xv.y);
    xo.z = f32_to_bf16_rne(xv.z); xo.w = f32_to_bf16_rne(xv.w);
    ao.x = f32_to_bf16_rne(av.x * inv); ao.y = f32_to_bf16_rne(av.y * inv);
    ao.z = f32_to_bf16_rne(av.z * inv); ao.w = f32_to_bf16_rne(av.w * inv);
    *(ushort4*)(xb + i4)   = xo;
    *(ushort4*)(aggb + i4) = ao;
}

// ================= LDS-tiled SAGEConv GEMM =================
// WG tile: 128 rows x 256 cols, K-step 32. 8 waves in a 2x4 grid; each wave 64x64.
// LDS rows padded to stride 40 (80 B) so 16-lane fragment reads start on 16
// distinct bank-quads (conflict-free ds_load_b128).
#define BM   128
#define BN   256
#define BK   32
#define LPAD 40    // padded row stride in bf16 elements

__global__ void __launch_bounds__(256)
sage_gemm_kernel(const unsigned short* __restrict__ xb, const unsigned short* __restrict__ aggb,
                 const unsigned short* __restrict__ wrT, const unsigned short* __restrict__ wnT,
                 const float* __restrict__ bias, const float* __restrict__ h1,
                 float* __restrict__ out, int n, int mode) {
    __shared__ unsigned short As[BM * LPAD];   // 10240 B
    __shared__ unsigned short Bs[BN * LPAD];   // 20480 B

    const int tid  = threadIdx.x;
    const int lane = tid & 31;
    const int wave = tid >> 5;
    const int waveM = wave >> 2;        // 0..1
    const int waveN = wave & 3;         // 0..3
    const int half  = lane >> 4;        // 0: lanes 0-15, 1: lanes 16-31
    const int l16   = lane & 15;

    const int rowTile = blockIdx.x * BM;
    const int colTile = blockIdx.y * BN;

    // ---- staging assignments ----
    // A: thread t loads row t/2, K-half (t&1)*16 (32 B); clamp row (select, no EXEC change)
    int arow = rowTile + (tid >> 1);
    if (arow >= n) arow = n - 1;
    const int aHalf = (tid & 1) * 16;
    const int aLds  = (tid >> 1) * LPAD + aHalf;           // element offset in As
    // B: thread t loads WT row (colTile + t), all 32 K (64 B)
    const int bLds  = tid * LPAD;

    // ---- fragment LDS offsets (elements) ----
    int aOff[4], bOff[4];
#pragma unroll
    for (int i = 0; i < 4; ++i) {
        aOff[i] = (waveM * 64 + i * 16 + l16) * LPAD + half * 8;
        bOff[i] = (waveN * 64 + i * 16 + l16) * LPAD + half * 8;
    }

    v8f acc[16];
#pragma unroll
    for (int i = 0; i < 16; ++i) acc[i] = (v8f){};

    union Frag { uint4 u[2]; v16bf v; };
    const int steps = DD / BK;   // 64

#pragma unroll 1
    for (int phase = 0; phase < 2; ++phase) {
        const unsigned short* Aglob = phase ? aggb : xb;
        const unsigned short* Bglob = phase ? wnT : wrT;
        const unsigned short* aBase = Aglob + (size_t)arow * DD + aHalf;
        const unsigned short* bBase = Bglob + (size_t)(colTile + tid) * DD;

        // preload tile 0 into registers
        uint4 aReg0 = *(const uint4*)(aBase);
        uint4 aReg1 = *(const uint4*)(aBase + 8);
        uint4 bReg0 = *(const uint4*)(bBase);
        uint4 bReg1 = *(const uint4*)(bBase + 8);
        uint4 bReg2 = *(const uint4*)(bBase + 16);
        uint4 bReg3 = *(const uint4*)(bBase + 24);

#pragma unroll 1
        for (int s = 0; s < steps; ++s) {
            // commit staged registers to LDS
            *(uint4*)(As + aLds)      = aReg0;
            *(uint4*)(As + aLds + 8)  = aReg1;
            *(uint4*)(Bs + bLds)      = bReg0;
            *(uint4*)(Bs + bLds + 8)  = bReg1;
            *(uint4*)(Bs + bLds + 16) = bReg2;
            *(uint4*)(Bs + bLds + 24) = bReg3;
            __syncthreads();

            // issue global loads for next tile (uniform branch)
            if (s + 1 < steps) {
                const int k = (s + 1) * BK;
                __builtin_prefetch(bBase + k + 256, 0, 1);
                aReg0 = *(const uint4*)(aBase + k);
                aReg1 = *(const uint4*)(aBase + k + 8);
                bReg0 = *(const uint4*)(bBase + k);
                bReg1 = *(const uint4*)(bBase + k + 8);
                bReg2 = *(const uint4*)(bBase + k + 16);
                bReg3 = *(const uint4*)(bBase + k + 24);
            }

            // B fragments resident; A fragment reloaded per mi (16 ds_load_b128 : 16 wmma)
            Frag Bf[4];
#pragma unroll
            for (int ni = 0; ni < 4; ++ni) {
                Bf[ni].u[0] = *(const uint4*)(Bs + bOff[ni]);
                Bf[ni].u[1] = *(const uint4*)(Bs + bOff[ni] + 16);
            }
#pragma unroll
            for (int mi = 0; mi < 4; ++mi) {
                Frag Af;
                Af.u[0] = *(const uint4*)(As + aOff[mi]);        // K = half*8 .. +7
                Af.u[1] = *(const uint4*)(As + aOff[mi] + 16);   // K = 16+half*8 .. +7
#pragma unroll
                for (int ni = 0; ni < 4; ++ni) {
                    acc[mi * 4 + ni] = __builtin_amdgcn_wmma_f32_16x16x32_bf16(
                        false, Af.v, false, Bf[ni].v, (short)0, acc[mi * 4 + ni], false, false);
                }
            }
            __syncthreads();
        }
    }

    // ---- epilogue: bias + ReLU / 0.5*(h1 + h2) ----
    float bv[4];
#pragma unroll
    for (int ni = 0; ni < 4; ++ni)
        bv[ni] = bias[colTile + waveN * 64 + ni * 16 + l16];

#pragma unroll
    for (int mi = 0; mi < 4; ++mi) {
#pragma unroll
        for (int ni = 0; ni < 4; ++ni) {
            const int colB = colTile + waveN * 64 + ni * 16 + l16;
            v8f a = acc[mi * 4 + ni];
#pragma unroll
            for (int v = 0; v < 8; ++v) {
                int r = rowTile + waveM * 64 + mi * 16 + v + 8 * half;  // C layout: M = vgpr + 8*(lane/16)
                if (r < n) {
                    size_t oi = (size_t)r * DD + colB;
                    float val = a[v] + bv[ni];
                    if (mode == 0) out[oi] = fmaxf(val, 0.0f);          // ReLU
                    else           out[oi] = 0.5f * (h1[oi] + val);     // 0.5*(h1+h2)
                }
            }
        }
    }
}

// ---------- host ----------
extern "C" void kernel_launch(void* const* d_in, const int* in_sizes, int n_in,
                              void* d_out, int out_size, void* d_ws, size_t ws_size,
                              hipStream_t stream) {
    const float* x    = (const float*)d_in[0];
    const int*   ei   = (const int*)d_in[1];
    const float* Wr   = (const float*)d_in[2];
    const float* Wn   = (const float*)d_in[3];
    const float* bias = (const float*)d_in[4];

    const int N = in_sizes[0] / DD;
    const int E = in_sizes[1] / 2;
    const int* src = ei;
    const int* dst = ei + E;

    // workspace layout
    char* ws = (char*)d_ws;
    unsigned short* wt = (unsigned short*)ws;                  // 8 * D*D bf16 (WrT/WnT interleaved)
    size_t wtBytes = (size_t)8 * DD * DD * sizeof(unsigned short);
    float* cur  = (float*)(ws + wtBytes);                      // N*D f32
    float* h1   = cur + (size_t)N * DD;                        // N*D f32
    float* agg  = h1  + (size_t)N * DD;                        // N*D f32
    unsigned short* xb   = (unsigned short*)(agg + (size_t)N * DD);  // N*D bf16
    unsigned short* aggb = xb + (size_t)N * DD;                      // N*D bf16
    float* deg = (float*)(aggb + (size_t)N * DD);              // N f32

    // degrees (once — same graph for all convs)
    hipMemsetAsync(deg, 0, (size_t)N * sizeof(float), stream);
    deg_kernel<<<(E + 255) / 256, 256, 0, stream>>>(dst, deg, E);

    // bf16 transposed weights (every call: deterministic, no caching)
    {
        long tot = 8L * DD * DD;
        wconv_kernel<<<(int)((tot + 255) / 256), 256, 0, stream>>>(Wr, Wn, wt);
    }

    dim3 ggrid((N + BM - 1) / BM, DD / BN);
    const long nd4 = (long)N * DD / 4;
    const long sth = (long)E * (DD / 4);

    for (int c = 0; c < 4; ++c) {
        // conv inputs: c0 <- x, c1 <- h1, c2 <- cur, c3 <- h1
        const float* in_f = (c == 0) ? x : ((c == 2) ? cur : h1);
        // conv outputs: c0 -> h1 (relu), c1 -> cur (combine), c2 -> h1 (relu), c3 -> d_out (combine)
        float* outp = (c == 1) ? cur : ((c == 3) ? (float*)d_out : h1);
        int mode = c & 1;

        hipMemsetAsync(agg, 0, (size_t)N * DD * sizeof(float), stream);
        scatter_kernel<<<(int)((sth + 255) / 256), 256, 0, stream>>>(in_f, src, dst, agg, E);
        normconv_kernel<<<(int)((nd4 + 255) / 256), 256, 0, stream>>>(in_f, agg, deg, xb, aggb, N);
        sage_gemm_kernel<<<ggrid, 256, 0, stream>>>(
            xb, aggb,
            wt + (size_t)(2 * c + 0) * DD * DD,
            wt + (size_t)(2 * c + 1) * DD * DD,
            bias + (size_t)c * DD, h1, outp, N, mode);
    }
}
